// MultiHeadAttention_36704790512012
// MI455X (gfx1250) — matrix operand order repaired and checked by reading, not compiled, not run on hardware
//
#include <hip/hip_runtime.h>

#define BATCH  2
#define SEQ    2048
#define DMODEL 768
#define NHEAD  12
#define DHEAD  64

typedef __attribute__((ext_vector_type(16))) _Float16 v16h;
typedef __attribute__((ext_vector_type(8)))  _Float16 v8h;
typedef __attribute__((ext_vector_type(8)))  float    v8f;
typedef __attribute__((ext_vector_type(4)))  float    v4f;

static __device__ __forceinline__ v8f wmma_f16(v16h a, v16h b, v8f c) {
  // D = A(16x32 f16) * B(32x16 f16) + C(16x16 f32)
  return __builtin_amdgcn_wmma_f32_16x16x32_f16(false, a, false, b, (short)0, c,
                                                false, false);
}

// A fragment (16x32 f16), LDS tile stored [m][k] (k contiguous).
// ISA layout: lane m = lane&15, half = lane>>4; elements e<8 -> k = 8*half+e,
// e>=8 -> k = 16 + 8*half + (e-8). Two 16B LDS loads.
static __device__ __forceinline__ v16h load_frag_a(const _Float16* rowp, int half) {
  v8h lo = *(const v8h*)(rowp + 8 * half);
  v8h hi = *(const v8h*)(rowp + 16 + 8 * half);
  return __builtin_shufflevector(lo, hi, 0,1,2,3,4,5,6,7,8,9,10,11,12,13,14,15);
}

// B fragment (32x16 f16), LDS tile stored TRANSPOSED as [n][k] (k contiguous).
// ISA layout: lane n = lane&15, half = lane>>4; elements e<8 -> k = 8*half+e,
// e>=8 -> k = 16 + 8*half + (e-8).
static __device__ __forceinline__ v16h load_frag_b(const _Float16* rowp, int half) {
  v8h lo = *(const v8h*)(rowp + 8 * half);
  v8h hi = *(const v8h*)(rowp + 16 + 8 * half);
  return __builtin_shufflevector(lo, hi, 0,1,2,3,4,5,6,7,8,9,10,11,12,13,14,15);
}

// CDNA5 async global->LDS copy, 16 bytes per lane. Tracked by ASYNCcnt.
// LDS offset = low 32 bits of the generic pointer (ISA 10.2: LDS aperture
// addresses truncate to the wave-relative LDS byte offset).
static __device__ __forceinline__ void async_copy16(void* lds_ptr,
                                                    const void* gptr) {
  unsigned int lds_off = (unsigned int)(unsigned long long)lds_ptr;
  unsigned long long ga = (unsigned long long)gptr;
  asm volatile("global_load_async_to_lds_b128 %0, %1, off"
               :
               : "v"(lds_off), "v"(ga)
               : "memory");
}

// ---------------------------------------------------------------------------
// Projection GEMM: C[4096 x 768] = A[4096 x 768] * W[768 x 768] + bias
// mode 0: A=q  -> Qh f16 [B,H,L,64]
// mode 1: A=k  -> Kh f16 [B,H,L,64]
// mode 2: A=v  -> Vt f16 [B,H,64,L]   (transposed for attention B-frags)
// mode 3: A=Oh(f16) * Wo -> out f32 [B,L,768]
// Block: 256 threads (8 waves), tile 128(M) x 64(N), K-step 32.
// ---------------------------------------------------------------------------
#define PBM 128
#define PBN 64
#define PBK 32
#define LDSA 40   // halfs per row (padded: 32+8)
#define LDSB 40

__global__ __launch_bounds__(256, 1)
void mha_proj_kernel(const float* __restrict__ q, const float* __restrict__ k,
                     const float* __restrict__ v,
                     const float* __restrict__ Wq, const float* __restrict__ bq,
                     const float* __restrict__ Wk, const float* __restrict__ bk,
                     const float* __restrict__ Wv, const float* __restrict__ bv,
                     const float* __restrict__ Wo, const float* __restrict__ bo,
                     const _Float16* __restrict__ Oh,
                     _Float16* __restrict__ Qh, _Float16* __restrict__ Kh,
                     _Float16* __restrict__ Vt, float* __restrict__ out,
                     int modeBase) {
  const int mode = modeBase + blockIdx.z;
  const float* A32 = q;
  const float* W = Wq;
  const float* bias = bq;
  if (mode == 1) { A32 = k; W = Wk; bias = bk; }
  else if (mode == 2) { A32 = v; W = Wv; bias = bv; }
  else if (mode == 3) { W = Wo; bias = bo; }

  __shared__ _Float16 As[PBM * LDSA];
  __shared__ _Float16 Bs[PBN * LDSB];

  const int t = threadIdx.x;
  const int lane = t & 31;
  const int wv = t >> 5;
  const int mb = wv & 3;   // wave row block (32 rows)
  const int nb = wv >> 2;  // wave col block (32 cols)
  const int mtile = blockIdx.y;  // 0..31
  const int ntile = blockIdx.x;  // 0..11
  const int half = lane >> 4;
  const int ln = lane & 15;

  v8f acc[2][2] = {};

  for (int kt = 0; kt < DMODEL / PBK; ++kt) {
    __syncthreads();
    // ---- stage A tile (128 x 32) into LDS as f16 ----
    if (mode < 3) {
      for (int i = 0; i < 4; ++i) {
        int c = t + i * 256;              // 0..1023 chunks of 4 floats
        int row = c >> 3, cc = c & 7;
        v4f f = *(const v4f*)(A32 + (size_t)(mtile * PBM + row) * DMODEL +
                              kt * PBK + cc * 4);
        _Float16* dst = As + row * LDSA + cc * 4;
        dst[0] = (_Float16)f.x; dst[1] = (_Float16)f.y;
        dst[2] = (_Float16)f.z; dst[3] = (_Float16)f.w;
      }
    } else {
      for (int i = 0; i < 2; ++i) {
        int c = t + i * 256;              // 0..511 chunks of 8 halfs
        int row = c >> 2, cc = c & 3;
        *(v8h*)(As + row * LDSA + cc * 8) =
            *(const v8h*)(Oh + (size_t)(mtile * PBM + row) * DMODEL +
                          kt * PBK + cc * 8);
      }
    }
    // ---- stage B tile transposed: Bs[n][k] (64 x 32) ----
    {
      int n = t & 63;
      int k0 = (t >> 6) * 8;
      const float* wp = W + (size_t)(kt * PBK + k0) * DMODEL + ntile * PBN + n;
      _Float16* bp = Bs + n * LDSB + k0;
#pragma unroll
      for (int kk = 0; kk < 8; ++kk) bp[kk] = (_Float16)wp[(size_t)kk * DMODEL];
    }
    __syncthreads();

    // ---- 4 WMMAs per wave per K-step ----
    v16h bf0 = load_frag_b(Bs + (32 * nb + ln) * LDSB, half);
    v16h bf1 = load_frag_b(Bs + (32 * nb + 16 + ln) * LDSB, half);
#pragma unroll
    for (int i = 0; i < 2; ++i) {
      v16h a = load_frag_a(As + (32 * mb + 16 * i + ln) * LDSA, half);
      acc[i][0] = wmma_f16(a, bf0, acc[i][0]);
      acc[i][1] = wmma_f16(a, bf1, acc[i][1]);
    }
  }

  // ---- epilogue: bias + scatter store ----
#pragma unroll
  for (int i = 0; i < 2; ++i) {
#pragma unroll
    for (int j = 0; j < 2; ++j) {
      int gn = ntile * PBN + 32 * nb + 16 * j + ln;   // global column
      float bs = bias[gn];
      int gm0 = mtile * PBM + 32 * mb + 16 * i + 8 * half;
#pragma unroll
      for (int r = 0; r < 8; ++r) {
        float val = acc[i][j][r] + bs;
        int row = gm0 + r;               // 0..4095
        if (mode == 3) {
          out[(size_t)row * DMODEL + gn] = val;
        } else {
          int bidx = row >> 11;          // row / 2048
          int l = row & 2047;
          int h = gn >> 6, e = gn & 63;
          _Float16 hv = (_Float16)val;
          if (mode == 0)
            Qh[(((size_t)bidx * NHEAD + h) * SEQ + l) * DHEAD + e] = hv;
          else if (mode == 1)
            Kh[(((size_t)bidx * NHEAD + h) * SEQ + l) * DHEAD + e] = hv;
          else
            Vt[(((size_t)bidx * NHEAD + h) * DHEAD + e) * SEQ + l] = hv;
        }
      }
    }
  }
}

// ---------------------------------------------------------------------------
// Flash attention: one block per (b, h, 128-query tile). 8 waves, each wave
// owns 16 full query rows (softmax reductions are width-16 shuffles).
// K/V tiles are double-buffered via GLOBAL_LOAD_ASYNC_TO_LDS_B128 + ASYNCcnt.
// Online softmax, O accumulated in f32, P re-laid-out via per-wave LDS tile.
// ---------------------------------------------------------------------------
#define ALD 72        // padded LDS row stride (halfs) for 64-wide tiles
#define QROWS 128     // query rows per block
#define NKT (SEQ / 64)

__global__ __launch_bounds__(256, 1)
void mha_attn_kernel(const _Float16* __restrict__ Qh,
                     const _Float16* __restrict__ Kh,
                     const _Float16* __restrict__ Vt,
                     const float* __restrict__ att_mask,
                     _Float16* __restrict__ Oh) {
  const int qt = blockIdx.x;   // query tile (0..15)
  const int h  = blockIdx.y;   // head
  const int b  = blockIdx.z;   // batch
  const int t = threadIdx.x;
  const int lane = t & 31;
  const int w = t >> 5;        // wave 0..7 -> query rows 16w..16w+15
  const int half = lane >> 4;
  const int ln = lane & 15;

  __shared__ _Float16 Qs[QROWS * ALD];
  __shared__ _Float16 Ks[2][64 * ALD];
  __shared__ _Float16 Vs[2][64 * ALD];   // [feature][key]
  __shared__ _Float16 Ps[8][16 * ALD];   // per-wave P tile [16 rows][64 keys]

  const _Float16* Qg = Qh + (((size_t)b * NHEAD + h) * SEQ + qt * QROWS) * DHEAD;
  const _Float16* Kg = Kh + (((size_t)b * NHEAD + h) * SEQ) * DHEAD;
  const _Float16* Vg = Vt + (((size_t)b * NHEAD + h) * DHEAD) * SEQ;

  // ---- async-stage Q tile (128x64 f16): 1024 16B chunks, 4 per thread ----
#pragma unroll
  for (int i = 0; i < 4; ++i) {
    int c = t + i * 256;
    int row = c >> 3, cc = c & 7;
    async_copy16(Qs + row * ALD + cc * 8, Qg + row * DHEAD + cc * 8);
  }
  // ---- async-stage K/V tile 0 into buffer 0: 2+2 chunks per thread ----
#pragma unroll
  for (int i = 0; i < 2; ++i) {
    int c = t + i * 256;
    int row = c >> 3, cc = c & 7;
    async_copy16(Ks[0] + row * ALD + cc * 8,
                 Kg + (size_t)row * DHEAD + cc * 8);
    async_copy16(Vs[0] + row * ALD + cc * 8,
                 Vg + (size_t)row * SEQ + cc * 8);
  }

  float mrun[8], lrun[8];
#pragma unroll
  for (int r = 0; r < 8; ++r) { mrun[r] = -3.0e38f; lrun[r] = 0.0f; }
  v8f oacc[4] = {};
  const float scale = 0.125f;  // 1/sqrt(64)

  for (int jt = 0; jt < NKT; ++jt) {
    const int cur = jt & 1;
    const _Float16* Kc = Ks[cur];
    const _Float16* Vc = Vs[cur];
    // all waves finished reading buffer cur^1 (iteration jt-1)
    __syncthreads();
    if (jt + 1 < NKT) {
      // prefetch next tile into the other buffer while computing this one
      _Float16* Kn = Ks[cur ^ 1];
      _Float16* Vn = Vs[cur ^ 1];
#pragma unroll
      for (int i = 0; i < 2; ++i) {
        int c = t + i * 256;
        int row = c >> 3, cc = c & 7;
        async_copy16(Kn + row * ALD + cc * 8,
                     Kg + ((size_t)((jt + 1) * 64 + row)) * DHEAD + cc * 8);
        async_copy16(Vn + row * ALD + cc * 8,
                     Vg + (size_t)row * SEQ + (jt + 1) * 64 + cc * 8);
      }
      // async loads retire in order: <=4 outstanding => tile jt (and Q) landed
      asm volatile("s_wait_asynccnt 0x4" ::: "memory");
    } else {
      asm volatile("s_wait_asynccnt 0x0" ::: "memory");
    }
    __syncthreads();  // every wave's portion of tile jt is in LDS

    // ---- S = Q K^T : 4 n-frags x 2 k-steps = 8 WMMAs ----
    v8f s[4] = {};
#pragma unroll
    for (int ks = 0; ks < 2; ++ks) {
      v16h a = load_frag_a(Qs + (16 * w + ln) * ALD + ks * 32, half);
#pragma unroll
      for (int f = 0; f < 4; ++f) {
        v16h bf = load_frag_b(Kc + (16 * f + ln) * ALD + ks * 32, half);
        s[f] = wmma_f16(a, bf, s[f]);
      }
    }

    // additive mask per key column (column n = 16*f + ln within tile)
    float mk[4];
#pragma unroll
    for (int f = 0; f < 4; ++f) {
      float m = att_mask[(size_t)b * SEQ + jt * 64 + 16 * f + ln];
      mk[f] = (1.0f - m) * -99999.0f;
    }

    // ---- online softmax; each lane's (r) is row m = r + 8*half ----
#pragma unroll
    for (int r = 0; r < 8; ++r) {
      float mx = -3.0e38f;
#pragma unroll
      for (int f = 0; f < 4; ++f)
        mx = fmaxf(mx, s[f][r] * scale + mk[f]);
      for (int d = 1; d < 16; d <<= 1) mx = fmaxf(mx, __shfl_xor(mx, d, 16));
      float mnew = fmaxf(mrun[r], mx);
      float fct = __expf(mrun[r] - mnew);
      mrun[r] = mnew;
      lrun[r] *= fct;
#pragma unroll
      for (int f2 = 0; f2 < 4; ++f2) oacc[f2][r] *= fct;
      float rs = 0.0f;
#pragma unroll
      for (int f = 0; f < 4; ++f) {
        float p = __expf(s[f][r] * scale + mk[f] - mnew);
        rs += p;
        Ps[w][(r + 8 * half) * ALD + 16 * f + ln] = (_Float16)p;
      }
      for (int d = 1; d < 16; d <<= 1) rs += __shfl_xor(rs, d, 16);
      lrun[r] += rs;
    }

    // ---- O += P * V : A from Ps (D->A relayout via LDS), B from Vs ----
#pragma unroll
    for (int ks = 0; ks < 2; ++ks) {
      v16h a = load_frag_a(Ps[w] + ln * ALD + ks * 32, half);
#pragma unroll
      for (int f2 = 0; f2 < 4; ++f2) {
        v16h bf = load_frag_b(Vc + (16 * f2 + ln) * ALD + ks * 32, half);
        oacc[f2] = wmma_f16(a, bf, oacc[f2]);
      }
    }
  }

  // ---- normalize and store Oh [B, L, H*64] f16 ----
#pragma unroll
  for (int f2 = 0; f2 < 4; ++f2) {
#pragma unroll
    for (int r = 0; r < 8; ++r) {
      float val = oacc[f2][r] / lrun[r];
      int row = qt * QROWS + 16 * w + r + 8 * half;
      int col = h * DHEAD + 16 * f2 + ln;
      Oh[((size_t)b * SEQ + row) * DMODEL + col] = (_Float16)val;
    }
  }
}

// ---------------------------------------------------------------------------
extern "C" void kernel_launch(void* const* d_in, const int* in_sizes, int n_in,
                              void* d_out, int out_size, void* d_ws,
                              size_t ws_size, hipStream_t stream) {
  const float* q  = (const float*)d_in[0];
  const float* k  = (const float*)d_in[1];
  const float* v  = (const float*)d_in[2];
  const float* am = (const float*)d_in[3];
  const float* Wq = (const float*)d_in[4];
  const float* bq = (const float*)d_in[5];
  const float* Wk = (const float*)d_in[6];
  const float* bk = (const float*)d_in[7];
  const float* Wv = (const float*)d_in[8];
  const float* bv = (const float*)d_in[9];
  const float* Wo = (const float*)d_in[10];
  const float* bo = (const float*)d_in[11];
  float* out = (float*)d_out;

  const size_t HBUF = (size_t)BATCH * NHEAD * SEQ * DHEAD;  // 3,145,728 elems
  _Float16* Qh = (_Float16*)d_ws;
  _Float16* Kh = Qh + HBUF;
  _Float16* Vt = Kh + HBUF;
  _Float16* Oh = Vt + HBUF;   // [B, L, 768] f16

  dim3 blk(256);
  dim3 gProjQKV(DMODEL / PBN, (BATCH * SEQ) / PBM, 3);  // 12 x 32 x 3
  mha_proj_kernel<<<gProjQKV, blk, 0, stream>>>(
      q, k, v, Wq, bq, Wk, bk, Wv, bv, Wo, bo, Oh, Qh, Kh, Vt, out, 0);

  dim3 gAttn(SEQ / QROWS, NHEAD, BATCH);                // 16 x 12 x 2
  mha_attn_kernel<<<gAttn, blk, 0, stream>>>(Qh, Kh, Vt, am, Oh);

  dim3 gProjO(DMODEL / PBN, (BATCH * SEQ) / PBM, 1);    // 12 x 32
  mha_proj_kernel<<<gProjO, blk, 0, stream>>>(
      q, k, v, Wq, bq, Wk, bk, Wv, bv, Wo, bo, Oh, Qh, Kh, Vt, out, 3);
}
